// RWKV7TimeMix_67705864454480
// MI455X (gfx1250) — matrix operand order repaired and checked by reading, not compiled, not run on hardware
//
#include <hip/hip_runtime.h>

// ---------------------------------------------------------------------------
// RWKV7 time-mix for MI455X (gfx1250, wave32, WMMA + Tensor Data Mover).
// B=1, T=1024, C=2048, H=32, N=64.
// Big GEMMs: v_wmma_f32_16x16x32_bf16, tiles staged into LDS by TDM
// (tensor_load_to_lds), double-buffered. Weights pre-transposed to [N,K] so
// both A and B tiles are contiguous row-major 128x32 copies.
// ---------------------------------------------------------------------------

constexpr int Tt  = 1024;
constexpr int Cc  = 2048;
constexpr int Hh  = 32;
constexpr int Nn  = 64;
constexpr int DLl = 64;
constexpr int ALl = 64;
constexpr int GLl = 128;

#define USE_TDM 1

typedef __attribute__((ext_vector_type(16))) __bf16 v16bf;
typedef __attribute__((ext_vector_type(8)))  __bf16 v8bf;
typedef __attribute__((ext_vector_type(8)))  float  v8f;
typedef unsigned int u32x4 __attribute__((ext_vector_type(4)));
typedef int          i32x8 __attribute__((ext_vector_type(8)));
typedef int          i32x4 __attribute__((ext_vector_type(4)));

#define DEVFN __device__ __forceinline__

DEVFN float wave_sum32(float v) {
#pragma unroll
  for (int off = 16; off > 0; off >>= 1) v += __shfl_xor(v, off, 32);
  return v;
}

// ---------------------------------------------------------------------------
// fp32 [K,N] -> bf16 transposed [N,K]  (32x32 LDS-tiled transpose)
// grid (N/32, K/32), block (32,8)
// ---------------------------------------------------------------------------
__global__ __launch_bounds__(256) void transpose_cvt_kernel(
    const float* __restrict__ src, __bf16* __restrict__ dst) {
  __shared__ float tile[32][33];
  const int n0 = blockIdx.x * 32;
  const int k0 = blockIdx.y * 32;
#pragma unroll
  for (int i = 0; i < 4; ++i) {
    int k = k0 + threadIdx.y + i * 8;
    tile[threadIdx.y + i * 8][threadIdx.x] = src[(size_t)k * Cc + n0 + threadIdx.x];
  }
  __syncthreads();
#pragma unroll
  for (int i = 0; i < 4; ++i) {
    int n = n0 + threadIdx.y + i * 8;
    dst[(size_t)n * Cc + k0 + threadIdx.x] = (__bf16)tile[threadIdx.x][threadIdx.y + i * 8];
  }
}

// ---------------------------------------------------------------------------
// Time-shift + mixing -> bf16 activation matrices for the r/k/v GEMMs
// ---------------------------------------------------------------------------
__global__ __launch_bounds__(256) void mix_bf16_kernel(
    const float* __restrict__ x,
    const float* __restrict__ mr, const float* __restrict__ mk,
    const float* __restrict__ mv,
    __bf16* __restrict__ xr, __bf16* __restrict__ xk, __bf16* __restrict__ xv) {
  int idx = blockIdx.x * 256 + threadIdx.x;        // exact grid: T*C/256 blocks
  int t = idx / Cc, c = idx % Cc;
  float xc = x[idx];
  float xp = (t > 0) ? x[idx - Cc] : 0.f;
  float dx = xp - xc;
  xr[idx] = (__bf16)(xc + dx * mr[c]);
  xk[idx] = (__bf16)(xc + dx * mk[c]);
  xv[idx] = (__bf16)(xc + dx * mv[c]);
}

// ---------------------------------------------------------------------------
// bf16 WMMA GEMM: C[M,N] = A[M,K] * B^T[N,K]^T.  Block tile 128x128, 8 waves,
// wave tile 64x32 = 4x2 wmma accumulators.  Double-buffered LDS tiles staged
// by the Tensor Data Mover (wave 0 issues, all waves compute).
// M,N,K multiples of 128/128/32 (exact for all uses here).
// ---------------------------------------------------------------------------
__global__ __launch_bounds__(256) void gemm_bf16_wmma(
    const __bf16* __restrict__ A,   // [M,K] row-major
    const __bf16* __restrict__ BT,  // [N,K] row-major (pre-transposed weights)
    float* __restrict__ Co, int M, int Nd, int K) {
  __shared__ __attribute__((aligned(128))) __bf16 smA[2][128 * 32];  // 8KB x2
  __shared__ __attribute__((aligned(128))) __bf16 smB[2][128 * 32];  // 8KB x2

  const int tid  = threadIdx.x;
  const int lane = tid & 31;
  const int wave = tid >> 5;
  const int wm   = wave >> 2;                 // 0..1 -> M offset 64*wm
  const int wn   = wave & 3;                  // 0..3 -> N offset 32*wn
  const int bm   = blockIdx.y * 128;
  const int bn   = blockIdx.x * 128;

  v8f zero = 0.0f;
  v8f acc[4][2];
#pragma unroll
  for (int i = 0; i < 4; ++i)
#pragma unroll
    for (int j = 0; j < 2; ++j) acc[i][j] = zero;

  const int half = (lane < 16) ? 1 : 0;
  const int mrow = lane & 15;
  const int kba  = half ? 0 : 8;   // A frag: lanes 0-15 K{0..7,16..23}, 16-31 K{8..15,24..31}
  const int kbb  = half ? 0 : 8;   // B frag: lanes 0-15 K{0..7,16..23}, 16-31 K{8..15,24..31}

  // per-iteration compute on LDS buffer `buf`
  auto compute = [&](int buf) {
    v16bf af[4];
#pragma unroll
    for (int i = 0; i < 4; ++i) {
      const __bf16* p = &smA[buf][(wm * 64 + i * 16 + mrow) * 32];
      union { v16bf v; v8bf h[2]; } u;
      u.h[0] = *(const v8bf*)(p + kba);
      u.h[1] = *(const v8bf*)(p + kba + 16);
      af[i] = u.v;
    }
    v16bf bfr[2];
#pragma unroll
    for (int j = 0; j < 2; ++j) {
      const __bf16* p = &smB[buf][(wn * 32 + j * 16 + mrow) * 32];
      union { v16bf v; v8bf h[2]; } u;
      u.h[0] = *(const v8bf*)(p + kbb);
      u.h[1] = *(const v8bf*)(p + kbb + 16);
      bfr[j] = u.v;
    }
#pragma unroll
    for (int i = 0; i < 4; ++i)
#pragma unroll
      for (int j = 0; j < 2; ++j)
        acc[i][j] = __builtin_amdgcn_wmma_f32_16x16x32_bf16(
            false, af[i], false, bfr[j], (short)0, acc[i][j], false, false);
  };

#if USE_TDM
  // ---- Tensor DMA descriptors (ISA cdna5 ch8): 2D tile 128 x 32 bf16 ----
  auto make_g1 = [](int rows, int Kdim) -> i32x8 {
    i32x8 g;
    g[0] = 0x00010000;                                  // data_size=1 (2 bytes)
    g[1] = (int)(((unsigned)Kdim & 0xFFFFu) << 16);     // tensor_dim0[15:0]
    g[2] = (int)(((unsigned)Kdim >> 16) |               // tensor_dim0[31:16]
                 (((unsigned)rows & 0xFFFFu) << 16));   // tensor_dim1[15:0]
    g[3] = (int)(((unsigned)rows >> 16) | (32u << 16)); // tensor_dim1[31:16] | tile_dim0=32
    g[4] = 128;                                         // tile_dim1 = 128
    g[5] = Kdim;                                        // tensor_dim0_stride[31:0]
    g[6] = 0;
    g[7] = 0;
    return g;
  };
  auto make_g0 = [](unsigned lds_addr, unsigned long long gaddr) -> u32x4 {
    u32x4 g;
    g[0] = 1u;                                          // count=1, no gather
    g[1] = lds_addr;                                    // LDS byte address
    g[2] = (unsigned)gaddr;                             // global_addr[31:0]
    g[3] = ((unsigned)(gaddr >> 32) & 0x01FFFFFFu) |    // global_addr[56:32]
           (2u << 30);                                  // type = 2 (image)
    return g;
  };
  const i32x8 g1A = make_g1(M, K);
  const i32x8 g1B = make_g1(Nd, K);
  const i32x4 z4 = {0, 0, 0, 0};
  const i32x8 z8 = {0, 0, 0, 0, 0, 0, 0, 0};
  auto issue = [&](int buf, int kt) {
    unsigned ldsA = (unsigned)(size_t)(&smA[buf][0]);   // generic addr low 32 = LDS offset
    unsigned ldsB = (unsigned)(size_t)(&smB[buf][0]);
    unsigned long long ga = (unsigned long long)(size_t)A  + ((size_t)bm * K + kt) * 2;
    unsigned long long gb = (unsigned long long)(size_t)BT + ((size_t)bn * K + kt) * 2;
    __builtin_amdgcn_tensor_load_to_lds(make_g0(ldsA, ga), g1A, z4, z4, z8, 0);
    __builtin_amdgcn_tensor_load_to_lds(make_g0(ldsB, gb), g1B, z4, z4, z8, 0);
  };

  if (wave == 0) issue(0, 0);                 // prologue: tile 0 -> buf 0
  for (int kt = 0; kt < K; kt += 32) {
    const int buf = (kt >> 5) & 1;
    if (wave == 0) __builtin_amdgcn_s_wait_tensorcnt((short)0);
    __syncthreads();                          // tile `buf` visible to all waves
    if (wave == 0 && kt + 32 < K) issue(buf ^ 1, kt + 32);  // overlap next DMA
    compute(buf);
    __syncthreads();                          // done reading before buf reused
  }
#else
  // ---- fallback: vector global->reg->LDS staging, double-buffered ----
  const int srow = tid >> 1;                  // 128 rows, 2 threads/row
  const int scol = (tid & 1) << 4;            // 0 or 16
  const uint4* gA = (const uint4*)(A  + (size_t)(bm + srow) * K + scol);
  const uint4* gB = (const uint4*)(BT + (size_t)(bn + srow) * K + scol);
  uint4 ra0 = gA[0], ra1 = gA[1], rb0 = gB[0], rb1 = gB[1];
  *(uint4*)&smA[0][srow * 32 + scol]     = ra0;
  *(uint4*)&smA[0][srow * 32 + scol + 8] = ra1;
  *(uint4*)&smB[0][srow * 32 + scol]     = rb0;
  *(uint4*)&smB[0][srow * 32 + scol + 8] = rb1;
  for (int kt = 0; kt < K; kt += 32) {
    const int buf = (kt >> 5) & 1;
    __syncthreads();
    const bool more = (kt + 32 < K);
    if (more) {
      const uint4* nA = (const uint4*)(A  + (size_t)(bm + srow) * K + kt + 32 + scol);
      const uint4* nB = (const uint4*)(BT + (size_t)(bn + srow) * K + kt + 32 + scol);
      ra0 = nA[0]; ra1 = nA[1]; rb0 = nB[0]; rb1 = nB[1];
      __builtin_prefetch(A  + (size_t)(bm + srow) * K + kt + 64 + scol, 0, 3);
      __builtin_prefetch(BT + (size_t)(bn + srow) * K + kt + 64 + scol, 0, 3);
    }
    compute(buf);
    if (more) {
      *(uint4*)&smA[buf ^ 1][srow * 32 + scol]     = ra0;
      *(uint4*)&smA[buf ^ 1][srow * 32 + scol + 8] = ra1;
      *(uint4*)&smB[buf ^ 1][srow * 32 + scol]     = rb0;
      *(uint4*)&smB[buf ^ 1][srow * 32 + scol + 8] = rb1;
    }
    __syncthreads();
  }
#endif

  // ---- writeback: D layout lanes 0-15 rows 0..7, lanes 16-31 rows 8..15 ----
  const int rbase = bm + wm * 64 + ((lane < 16) ? 0 : 8);
  const int cbase = bn + wn * 32 + (lane & 15);
#pragma unroll
  for (int i = 0; i < 4; ++i)
#pragma unroll
    for (int j = 0; j < 2; ++j)
#pragma unroll
      for (int e = 0; e < 8; ++e)
        Co[(size_t)(rbase + i * 16 + e) * Nd + cbase + j * 16] = acc[i][j][e];
}

// ---------------------------------------------------------------------------
// LoRA hidden: hid[t,d] = act( sum_c (x + (shift(x)-x)*mix)[t,c] * W1[c,d] )
// mode: 0 = tanh, 1 = identity, 2 = sigmoid
// ---------------------------------------------------------------------------
__global__ __launch_bounds__(256) void lora_hidden_kernel(
    const float* __restrict__ x, const float* __restrict__ mix,
    const float* __restrict__ W1, float* __restrict__ hid, int D, int mode) {
  int idx = blockIdx.x * 256 + threadIdx.x;   // exact: T*D/256 blocks
  int t = idx / D, d = idx % D;
  const float* xr = x + (size_t)t * Cc;
  float s = 0.f;
  for (int c = 0; c < Cc; ++c) {
    float xc = xr[c];
    float xp = (t > 0) ? xr[c - Cc] : 0.f;
    float xm = xc + (xp - xc) * mix[c];
    s = fmaf(xm, W1[(size_t)c * D + d], s);
  }
  if (mode == 0)      s = tanhf(s);
  else if (mode == 2) s = 1.f / (1.f + expf(-s));
  hid[idx] = s;
}

// w path: w = -softplus(-(w0 + hid@w2)) - 0.5 ; store wdec = exp(-exp(w))
__global__ __launch_bounds__(256) void lora_w_out_kernel(
    const float* __restrict__ hid, const float* __restrict__ w2,
    const float* __restrict__ w0, float* __restrict__ wdec) {
  int idx = blockIdx.x * 256 + threadIdx.x;   // T*C
  int t = idx / Cc, c = idx % Cc;
  float s = 0.f;
#pragma unroll 8
  for (int d = 0; d < DLl; ++d) s = fmaf(hid[t * DLl + d], w2[(size_t)d * Cc + c], s);
  float y = w0[c] + s;
  float w = -log1pf(expf(-y)) - 0.5f;
  wdec[idx] = expf(-expf(w));
}

__global__ __launch_bounds__(256) void lora_a_out_kernel(
    const float* __restrict__ hid, const float* __restrict__ a2,
    const float* __restrict__ a0, float* __restrict__ aout) {
  int idx = blockIdx.x * 256 + threadIdx.x;
  int t = idx / Cc, c = idx % Cc;
  float s = 0.f;
#pragma unroll 8
  for (int d = 0; d < ALl; ++d) s = fmaf(hid[t * ALl + d], a2[(size_t)d * Cc + c], s);
  aout[idx] = 1.f / (1.f + expf(-(a0[c] + s)));
}

__global__ __launch_bounds__(256) void lora_g_out_kernel(
    const float* __restrict__ hid, const float* __restrict__ g2,
    float* __restrict__ gout) {
  int idx = blockIdx.x * 256 + threadIdx.x;
  int t = idx / Cc, c = idx % Cc;
  float s = 0.f;
#pragma unroll 8
  for (int d = 0; d < GLl; ++d) s = fmaf(hid[t * GLl + d], g2[(size_t)d * Cc + c], s);
  gout[idx] = s;
}

// ---------------------------------------------------------------------------
// Per-(t,head) prep: kk = normalize(k*k_k); aa=-kk; bb=kk*a; k <- k*(1+(a-1)*k_a)
// One wave32 per (t,h); each lane handles channels lane and lane+32.
// ---------------------------------------------------------------------------
__global__ __launch_bounds__(256) void prep_kernel(
    float* __restrict__ k, const float* __restrict__ a,
    const float* __restrict__ k_k, const float* __restrict__ k_a,
    float* __restrict__ aa, float* __restrict__ bb) {
  int gw   = blockIdx.x * 8 + (threadIdx.x >> 5);   // global wave, T*H total
  int lane = threadIdx.x & 31;
  int t = gw / Hh, h = gw % Hh;
  int base = t * Cc + h * Nn;
  int c0 = h * Nn + lane, c1 = c0 + 32;
  float kv0 = k[base + lane], kv1 = k[base + lane + 32];
  float kk0 = kv0 * k_k[c0], kk1 = kv1 * k_k[c1];
  float ss = wave_sum32(kk0 * kk0 + kk1 * kk1);
  float inv = 1.f / fmaxf(sqrtf(ss), 1e-12f);
  kk0 *= inv; kk1 *= inv;
  float av0 = a[base + lane], av1 = a[base + lane + 32];
  aa[base + lane]      = -kk0;
  aa[base + lane + 32] = -kk1;
  bb[base + lane]      = kk0 * av0;
  bb[base + lane + 32] = kk1 * av1;
  k[base + lane]       = kv0 * (1.f + (av0 - 1.f) * k_a[c0]);
  k[base + lane + 32]  = kv1 * (1.f + (av1 - 1.f) * k_a[c1]);
}

// ---------------------------------------------------------------------------
// Delta-rule recurrence: one block per head, 64 threads, each thread owns one
// state row (64 floats in VGPRs).  Per step: vectors broadcast via LDS.
// ---------------------------------------------------------------------------
__global__ __launch_bounds__(64) void recurrence_kernel(
    const float* __restrict__ r, const float* __restrict__ wdec,
    const float* __restrict__ k, const float* __restrict__ v,
    const float* __restrict__ aa, const float* __restrict__ bb,
    float* __restrict__ out) {
  __shared__ float rs[Nn], ws[Nn], ks[Nn], vs[Nn], as_[Nn], bs[Nn];
  const int h = blockIdx.x;
  const int i = threadIdx.x;
  const int hb = h * Nn;
  float st[Nn];
#pragma unroll
  for (int j = 0; j < Nn; ++j) st[j] = 0.f;

  for (int t = 0; t < Tt; ++t) {
    const int base = t * Cc + hb;
    rs[i]  = r[base + i];
    ws[i]  = wdec[base + i];
    ks[i]  = k[base + i];
    vs[i]  = v[base + i];
    as_[i] = aa[base + i];
    bs[i]  = bb[base + i];
    __syncthreads();
    float sa = 0.f;
#pragma unroll
    for (int j = 0; j < Nn; ++j) sa = fmaf(st[j], as_[j], sa);
    const float vi = vs[i];
    float o = 0.f;
#pragma unroll
    for (int j = 0; j < Nn; ++j) {
      st[j] = fmaf(st[j], ws[j], fmaf(sa, bs[j], vi * ks[j]));
      o = fmaf(st[j], rs[j], o);
    }
    out[base + i] = o;
    __syncthreads();
  }
}

// ---------------------------------------------------------------------------
// Per-head GroupNorm (eps=6.4e-4) + r_k bonus + gate -> bf16 for final GEMM.
// One wave32 per (t,h); lane handles channels lane and lane+32.
// ---------------------------------------------------------------------------
__global__ __launch_bounds__(256) void norm_gate_kernel(
    const float* __restrict__ out, const float* __restrict__ r,
    const float* __restrict__ k, const float* __restrict__ v,
    const float* __restrict__ r_k, const float* __restrict__ gamma,
    const float* __restrict__ beta, const float* __restrict__ g,
    __bf16* __restrict__ xog) {
  int gw   = blockIdx.x * 8 + (threadIdx.x >> 5);
  int lane = threadIdx.x & 31;
  int t = gw / Hh, h = gw % Hh;
  int base = t * Cc + h * Nn;
  int c0 = h * Nn + lane, c1 = c0 + 32;
  float o0 = out[base + lane], o1 = out[base + lane + 32];
  float s  = wave_sum32(o0 + o1);
  float sq = wave_sum32(o0 * o0 + o1 * o1);
  float mu  = s * (1.f / 64.f);
  float var = sq * (1.f / 64.f) - mu * mu;
  float inv = rsqrtf(var + 6.4e-4f);
  float bn  = wave_sum32(r[base + lane] * k[base + lane] * r_k[c0] +
                         r[base + lane + 32] * k[base + lane + 32] * r_k[c1]);
  float xo0 = (o0 - mu) * inv * gamma[c0] + beta[c0] + bn * v[base + lane];
  float xo1 = (o1 - mu) * inv * gamma[c1] + beta[c1] + bn * v[base + lane + 32];
  xog[base + lane]      = (__bf16)(xo0 * g[base + lane]);
  xog[base + lane + 32] = (__bf16)(xo1 * g[base + lane + 32]);
}

// ---------------------------------------------------------------------------
// Host-side launch
// ---------------------------------------------------------------------------
extern "C" void kernel_launch(void* const* d_in, const int* in_sizes, int n_in,
                              void* d_out, int out_size, void* d_ws, size_t ws_size,
                              hipStream_t stream) {
  (void)in_sizes; (void)n_in; (void)out_size; (void)ws_size;
  const float* x    = (const float*)d_in[0];
  const float* x_r  = (const float*)d_in[1];
  const float* x_w  = (const float*)d_in[2];
  const float* x_k  = (const float*)d_in[3];
  const float* x_v  = (const float*)d_in[4];
  const float* x_a  = (const float*)d_in[5];
  const float* x_g  = (const float*)d_in[6];
  const float* w0   = (const float*)d_in[7];
  const float* a0   = (const float*)d_in[8];
  const float* k_k  = (const float*)d_in[10];
  const float* k_a  = (const float*)d_in[11];
  const float* ln_g = (const float*)d_in[12];
  const float* ln_b = (const float*)d_in[13];
  const float* w1   = (const float*)d_in[14];
  const float* w2   = (const float*)d_in[15];
  const float* a1   = (const float*)d_in[16];
  const float* a2   = (const float*)d_in[17];
  const float* g1   = (const float*)d_in[20];
  const float* g2   = (const float*)d_in[21];
  const float* r_k  = (const float*)d_in[22];
  const float* W_r  = (const float*)d_in[23];
  const float* W_k  = (const float*)d_in[24];
  const float* W_v  = (const float*)d_in[25];
  const float* W_o  = (const float*)d_in[26];
  float* outp = (float*)d_out;

  const size_t TC = (size_t)Tt * Cc;          // 2,097,152
  const size_t CC = (size_t)Cc * Cc;          // 4,194,304
  size_t off = 0;
  auto carve = [&](size_t bytes) -> char* {
    char* p = (char*)d_ws + off;
    off += (bytes + 255) & ~(size_t)255;
    return p;
  };
  __bf16* WrT  = (__bf16*)carve(CC * 2);      // transposed bf16 weights [N,K]
  __bf16* WkT  = (__bf16*)carve(CC * 2);
  __bf16* WvT  = (__bf16*)carve(CC * 2);
  __bf16* WoT  = (__bf16*)carve(CC * 2);
  __bf16* xrB  = (__bf16*)carve(TC * 2);
  __bf16* xkB  = (__bf16*)carve(TC * 2);
  __bf16* xvB  = (__bf16*)carve(TC * 2);
  float*  rbuf = (float*)carve(TC * 4);
  float*  kbuf = (float*)carve(TC * 4);
  float*  vbuf = (float*)carve(TC * 4);
  float*  wdec = (float*)carve(TC * 4);
  float*  abuf = (float*)carve(TC * 4);
  float*  gbuf = (float*)carve(TC * 4);
  float*  aabf = (float*)carve(TC * 4);
  float*  bbbf = (float*)carve(TC * 4);
  float*  obuf = (float*)carve(TC * 4);
  float*  hidW = (float*)carve((size_t)Tt * DLl * 4);
  float*  hidA = (float*)carve((size_t)Tt * ALl * 4);
  float*  hidG = (float*)carve((size_t)Tt * GLl * 4);
  __bf16* xogB = (__bf16*)carve(TC * 2);

  dim3 tgrid(Cc / 32, Cc / 32);               // (64, 64)
  dim3 tblk(32, 8);
  transpose_cvt_kernel<<<tgrid, tblk, 0, stream>>>(W_r, WrT);
  transpose_cvt_kernel<<<tgrid, tblk, 0, stream>>>(W_k, WkT);
  transpose_cvt_kernel<<<tgrid, tblk, 0, stream>>>(W_v, WvT);
  transpose_cvt_kernel<<<tgrid, tblk, 0, stream>>>(W_o, WoT);

  const int ewBlocks = (int)(TC / 256);       // 8192
  mix_bf16_kernel<<<ewBlocks, 256, 0, stream>>>(x, x_r, x_k, x_v, xrB, xkB, xvB);

  dim3 ggrid(Cc / 128, Tt / 128);             // (16, 8)
  gemm_bf16_wmma<<<ggrid, 256, 0, stream>>>(xrB, WrT, rbuf, Tt, Cc, Cc);
  gemm_bf16_wmma<<<ggrid, 256, 0, stream>>>(xkB, WkT, kbuf, Tt, Cc, Cc);
  gemm_bf16_wmma<<<ggrid, 256, 0, stream>>>(xvB, WvT, vbuf, Tt, Cc, Cc);

  lora_hidden_kernel<<<(Tt * DLl) / 256, 256, 0, stream>>>(x, x_w, w1, hidW, DLl, 0);
  lora_hidden_kernel<<<(Tt * ALl) / 256, 256, 0, stream>>>(x, x_a, a1, hidA, ALl, 1);
  lora_hidden_kernel<<<(Tt * GLl) / 256, 256, 0, stream>>>(x, x_g, g1, hidG, GLl, 2);
  lora_w_out_kernel<<<ewBlocks, 256, 0, stream>>>(hidW, w2, w0, wdec);
  lora_a_out_kernel<<<ewBlocks, 256, 0, stream>>>(hidA, a2, a0, abuf);
  lora_g_out_kernel<<<ewBlocks, 256, 0, stream>>>(hidG, g2, gbuf);

  const int whBlocks = (Tt * Hh) / 8;         // 4096 blocks of 8 waves
  prep_kernel<<<whBlocks, 256, 0, stream>>>(kbuf, abuf, k_k, k_a, aabf, bbbf);

  recurrence_kernel<<<Hh, Nn, 0, stream>>>(rbuf, wdec, kbuf, vbuf, aabf, bbbf, obuf);

  norm_gate_kernel<<<whBlocks, 256, 0, stream>>>(obuf, rbuf, kbuf, vbuf, r_k,
                                                 ln_g, ln_b, gbuf, xogB);

  gemm_bf16_wmma<<<ggrid, 256, 0, stream>>>(xogB, WoT, outp, Tt, Cc, Cc);
}
